// LSTM_66443144069713
// MI455X (gfx1250) — matrix operand order repaired and checked by reading, not compiled, not run on hardware
//
#include <hip/hip_runtime.h>
#include <hip/hip_bf16.h>
#include <stdint.h>

#define E_  128
#define H_  192
#define L_  3
#define B_  256
#define T_  512
#define G4H (4 * H_)      // 768
#define NWG 12            // 192 h-cols / 16 per WG
#define ACT_STRIDE 40     // 32 halves + 8 pad -> 80B rows, 16B aligned for async b128 DMA
#define ACT_BUF_HALVES (B_ * ACT_STRIDE)

typedef _Float16 h16;
typedef __attribute__((ext_vector_type(2)))  _Float16 v2h;
typedef __attribute__((ext_vector_type(16))) _Float16 v16h;
typedef __attribute__((ext_vector_type(8)))  float    v8f;

__device__ __forceinline__ float fsig(float x)  { return 1.f / (1.f + __expf(-x)); }
__device__ __forceinline__ float ftanh(float x) { return 2.f / (1.f + __expf(-2.f * x)) - 1.f; }

// A fragment: 16x32 f16 (M x K). Lanes 0-15: M=lane, K groups {0..7,16..23};
// lanes 16-31: M=lane-16, K groups {8..15,24..31}. 2 halves per VGPR, 8 VGPRs.
__device__ __forceinline__ v16h load_a_frag(const h16* act, int rowBase, int lane) {
  const h16* p = act + (rowBase + (lane & 15)) * ACT_STRIDE + ((lane >> 4) << 3);
  v16h a;
#pragma unroll
  for (int v = 0; v < 8; ++v) {
    int k = ((v & 3) << 1) + ((v >> 2) << 4);
    v2h t = *(const v2h*)(p + k);
    a[2 * v]     = t[0];
    a[2 * v + 1] = t[1];
  }
  return a;
}

// B fragment: 32x16 f16 (K x N) from LDS weights stored transposed [col][k].
// Lanes 0-15: N=lane, K groups {0..7,16..23}; lanes 16-31: N=lane-16, K groups {8..15,24..31}.
__device__ __forceinline__ v16h load_b_frag(const h16* w, int colBase, int kOff, int kPad, int lane) {
  const h16* p = w + (colBase + (lane & 15)) * kPad + kOff + ((lane >> 4) << 3);
  v16h b;
#pragma unroll
  for (int v = 0; v < 8; ++v) {
    v2h t = *(const v2h*)(p + ((v & 3) << 1) + ((v >> 2) << 4));
    b[2 * v]     = t[0];
    b[2 * v + 1] = t[1];
  }
  return b;
}

// Per-direction grid barrier: monotonic counter, reset each launch (graph-safe).
__device__ __forceinline__ void grid_barrier(unsigned* cnt, unsigned target) {
  __threadfence();
  __syncthreads();
  if (threadIdx.x == 0) {
    __hip_atomic_fetch_add(cnt, 1u, __ATOMIC_RELEASE, __HIP_MEMORY_SCOPE_AGENT);
    while (__hip_atomic_load(cnt, __ATOMIC_ACQUIRE, __HIP_MEMORY_SCOPE_AGENT) < target)
      __builtin_amdgcn_s_sleep(1);
  }
  __syncthreads();
}

__global__ void reset_kernel(h16* Hbuf, unsigned* cnts) {
  int i = blockIdx.x * blockDim.x + threadIdx.x;
  const int nH = L_ * 2 * 2 * B_ * H_;
  if (i < nH) Hbuf[i] = (h16)0.f;
  if (i < 6 * 32) cnts[i] = 0u;
}

__global__ void embed_kernel(const int* __restrict__ ids, const float* __restrict__ emb,
                             h16* __restrict__ X) {
  int bt = blockIdx.x;       // b*T + t
  int e  = threadIdx.x;      // 0..127
  int id = ids[bt];
  X[(size_t)bt * E_ + e] = (h16)emb[(size_t)id * E_ + e];
}

// One bidirectional LSTM layer. grid = (NWG, 2 dirs), block = 256 (8 waves).
// Each WG owns 16 h-columns -> 64 z-columns {i,f,g,o}; each wave owns 32 batch rows.
// Activation K-chunks are double-buffered in LDS and filled by the CDNA5 async
// global->LDS DMA path (ASYNCcnt), overlapping DMA of chunk c+1 with WMMA of chunk c.
template <int I>
__global__ __launch_bounds__(256, 1) void lstm_layer_kernel(
    const h16* __restrict__ Xin,   // [B, T, I] f16
    h16* __restrict__ Yout,        // [B, T, 2H] f16
    const float* __restrict__ Wih, // [2, 4H, I]
    const float* __restrict__ Whh, // [2, 4H, H]
    const float* __restrict__ bias,// [2, 4H]
    h16* __restrict__ HbufL,       // [2 dirs][2 pp][B][H] f16
    unsigned* __restrict__ cnts)   // per-dir counters (stride 32 uints)
{
  constexpr int KT   = I + H_;      // concat K
  constexpr int KPAD = KT + 2;      // odd-dword weight rows
  constexpr int NK   = KT / 32;
  static_assert(KT % 32 == 0, "K must be chunkable by 32");

  extern __shared__ char smem[];
  h16* wLDS   = (h16*)smem;                          // [64][KPAD] resident weights
  h16* actLDS = (h16*)(smem + (size_t)64 * KPAD * 2);// 2 x [256][ACT_STRIDE] chunks

  const int d    = blockIdx.y;
  const int n0   = blockIdx.x * 16;
  const int lane = threadIdx.x & 31;
  const int wrow = (threadIdx.x >> 5) * 32;

  const float* WihD = Wih + (size_t)d * G4H * I;
  const float* WhhD = Whh + (size_t)d * G4H * H_;
  const float* bD   = bias + d * G4H;
  h16* Hb = HbufL + (size_t)d * 2 * B_ * H_;
  unsigned* cnt = cnts + d * 32;

  // Stage weights into LDS, transposed: wLDS[c][k] = W[gate*H + hcol][k]
  for (int idx = threadIdx.x; idx < 64 * KT; idx += 256) {
    int c = idx / KT, k = idx - c * KT;
    int row = ((c >> 4) * H_) + n0 + (c & 15);
    float wv = (k < I) ? WihD[(size_t)row * I + k] : WhhD[(size_t)row * H_ + (k - I)];
    wLDS[c * KPAD + k] = (h16)wv;
  }

  const int   hcol = n0 + (lane & 15);
  const float bi  = bD[0 * H_ + hcol];
  const float bff = bD[1 * H_ + hcol];
  const float bg  = bD[2 * H_ + hcol];
  const float bo  = bD[3 * H_ + hcol];
  const int   mh  = (lane >> 4) << 3;

  float creg[2][8];   // cell state lives in registers for all T steps
#pragma unroll
  for (int a = 0; a < 2; ++a)
#pragma unroll
    for (int r = 0; r < 8; ++r) creg[a][r] = 0.f;

  __syncthreads();

  for (int t = 0; t < T_; ++t) {
    const int tt = d ? (T_ - 1 - t) : t;
    const int p  = t & 1;

    // Async-DMA one [256 x 32] activation chunk (64B/row) into LDS buffer bufSel.
    auto stage_async = [&](int chunk, int bufSel) {
      const int b  = threadIdx.x;
      const int k0 = chunk * 32;
      const h16* src = (k0 < I)
          ? (Xin + ((size_t)b * T_ + tt) * I + k0)
          : (Hb + (size_t)p * B_ * H_ + (size_t)b * H_ + (k0 - I));
      uint32_t dst = (uint32_t)(uintptr_t)(actLDS + bufSel * ACT_BUF_HALVES + b * ACT_STRIDE);
      unsigned long long ga = (unsigned long long)(uintptr_t)src;
      asm volatile(
          "global_load_async_to_lds_b128 %0, %1, off\n\t"
          "global_load_async_to_lds_b128 %0, %1, off offset:16\n\t"
          "global_load_async_to_lds_b128 %0, %1, off offset:32\n\t"
          "global_load_async_to_lds_b128 %0, %1, off offset:48"
          :: "v"(dst), "v"(ga) : "memory");
    };

    v8f acc[2][4];
#pragma unroll
    for (int a = 0; a < 2; ++a)
#pragma unroll
      for (int g = 0; g < 4; ++g) acc[a][g] = v8f{0.f, 0.f, 0.f, 0.f, 0.f, 0.f, 0.f, 0.f};

    stage_async(0, 0);   // buf0 free: protected by grid_barrier of previous step

    for (int chunk = 0; chunk < NK; ++chunk) {
      // Wait for this wave's DMA of chunk `chunk`, then block-wide sync so every
      // thread's chunk data is visible; then kick off chunk+1 and compute.
      asm volatile("s_wait_asynccnt 0x0" ::: "memory");
      __syncthreads();
      if (chunk + 1 < NK) stage_async(chunk + 1, (chunk + 1) & 1);

      const h16* act = actLDS + (chunk & 1) * ACT_BUF_HALVES;
      v16h a0 = load_a_frag(act, wrow, lane);
      v16h a1 = load_a_frag(act, wrow + 16, lane);
      v16h bf0 = load_b_frag(wLDS, 0,  chunk * 32, KPAD, lane);
      v16h bf1 = load_b_frag(wLDS, 16, chunk * 32, KPAD, lane);
      v16h bf2 = load_b_frag(wLDS, 32, chunk * 32, KPAD, lane);
      v16h bf3 = load_b_frag(wLDS, 48, chunk * 32, KPAD, lane);
      acc[0][0] = __builtin_amdgcn_wmma_f32_16x16x32_f16(false, a0, false, bf0, (short)0, acc[0][0], false, false);
      acc[1][0] = __builtin_amdgcn_wmma_f32_16x16x32_f16(false, a1, false, bf0, (short)0, acc[1][0], false, false);
      acc[0][1] = __builtin_amdgcn_wmma_f32_16x16x32_f16(false, a0, false, bf1, (short)0, acc[0][1], false, false);
      acc[1][1] = __builtin_amdgcn_wmma_f32_16x16x32_f16(false, a1, false, bf1, (short)0, acc[1][1], false, false);
      acc[0][2] = __builtin_amdgcn_wmma_f32_16x16x32_f16(false, a0, false, bf2, (short)0, acc[0][2], false, false);
      acc[1][2] = __builtin_amdgcn_wmma_f32_16x16x32_f16(false, a1, false, bf2, (short)0, acc[1][2], false, false);
      acc[0][3] = __builtin_amdgcn_wmma_f32_16x16x32_f16(false, a0, false, bf3, (short)0, acc[0][3], false, false);
      acc[1][3] = __builtin_amdgcn_wmma_f32_16x16x32_f16(false, a1, false, bf3, (short)0, acc[1][3], false, false);
    }

    // Gate nonlinearities directly on the C/D layout: element r -> M = base+r(+8 hi), N = lane%16
    h16* HbN = Hb + (size_t)(p ^ 1) * B_ * H_;
#pragma unroll
    for (int rb = 0; rb < 2; ++rb) {
      const int mbase = wrow + rb * 16 + mh;
#pragma unroll
      for (int r = 0; r < 8; ++r) {
        const int m  = mbase + r;
        float zi = acc[rb][0][r] + bi;
        float zf = acc[rb][1][r] + bff;
        float zg = acc[rb][2][r] + bg;
        float zo = acc[rb][3][r] + bo;
        float cN = fsig(zf) * creg[rb][r] + fsig(zi) * ftanh(zg);
        float hN = fsig(zo) * ftanh(cN);
        creg[rb][r] = cN;
        h16 hv = (h16)hN;
        HbN[(size_t)m * H_ + hcol] = hv;
        Yout[((size_t)m * T_ + tt) * (2 * H_) + d * H_ + hcol] = hv;
      }
    }

    grid_barrier(cnt, (unsigned)(NWG * (t + 1)));
  }
}

__global__ void fc_kernel(const h16* __restrict__ Y, const float* __restrict__ w,
                          const float* __restrict__ fb, float* __restrict__ out) {
  int b = threadIdx.x;
  if (b < B_) {
    float s = fb[0];
    const h16* hf  = Y + ((size_t)b * T_ + (T_ - 1)) * (2 * H_);        // hT fwd
    const h16* hbk = Y + ((size_t)b * T_) * (2 * H_) + H_;              // hT bwd (t=0)
    for (int j = 0; j < H_; ++j) s += (float)hf[j] * w[j];
    for (int j = 0; j < H_; ++j) s += (float)hbk[j] * w[H_ + j];
    out[b] = s;
  }
}

extern "C" void kernel_launch(void* const* d_in, const int* in_sizes, int n_in,
                              void* d_out, int out_size, void* d_ws, size_t ws_size,
                              hipStream_t stream) {
  const int*   ids  = (const int*)  d_in[0];
  const float* emb  = (const float*)d_in[1];
  const float* Wih0 = (const float*)d_in[2];
  const float* Whh0 = (const float*)d_in[3];
  const float* b0   = (const float*)d_in[4];
  const float* WihR = (const float*)d_in[5];
  const float* WhhR = (const float*)d_in[6];
  const float* bR   = (const float*)d_in[7];
  const float* fcw  = (const float*)d_in[8];
  const float* fcb  = (const float*)d_in[9];
  float* out = (float*)d_out;

  char* ws = (char*)d_ws;
  const size_t szX0 = (size_t)B_ * T_ * E_ * 2;       // 33.5 MB
  const size_t szY  = (size_t)B_ * T_ * 2 * H_ * 2;   // 100.7 MB
  const size_t szHb = (size_t)L_ * 2 * 2 * B_ * H_ * 2;
  h16* X0   = (h16*)ws;
  h16* YA   = (h16*)(ws + szX0);
  h16* YB   = (h16*)(ws + szX0 + szY);
  h16* Hbuf = (h16*)(ws + szX0 + 2 * szY);
  unsigned* cnts = (unsigned*)(ws + szX0 + 2 * szY + szHb);

  {
    int nH = L_ * 2 * 2 * B_ * H_;
    reset_kernel<<<(nH + 255) / 256, 256, 0, stream>>>(Hbuf, cnts);
  }
  embed_kernel<<<B_ * T_, E_, 0, stream>>>(ids, emb, X0);

  const size_t actBytes = (size_t)2 * ACT_BUF_HALVES * 2;                      // 40 KB
  const size_t sh0 = (size_t)64 * (E_ + H_ + 2) * 2 + actBytes;                // ~80 KB
  const size_t sh1 = (size_t)64 * (2 * H_ + H_ + 2) * 2 + actBytes;            // ~112 KB

  // Layer 0: X0 -> YA
  lstm_layer_kernel<E_><<<dim3(NWG, 2), 256, sh0, stream>>>(
      X0, YA, Wih0, Whh0, b0, Hbuf, cnts);
  // Layer 1: YA -> YB
  lstm_layer_kernel<2 * H_><<<dim3(NWG, 2), 256, sh1, stream>>>(
      YA, YB, WihR, WhhR, bR,
      Hbuf + (size_t)1 * (2 * 2 * B_ * H_), cnts + 2 * 32);
  // Layer 2: YB -> YA
  lstm_layer_kernel<2 * H_><<<dim3(NWG, 2), 256, sh1, stream>>>(
      YB, YA,
      WihR + (size_t)2 * G4H * (2 * H_), WhhR + (size_t)2 * G4H * H_, bR + 2 * G4H,
      Hbuf + (size_t)2 * (2 * 2 * B_ * H_), cnts + 4 * 32);

  fc_kernel<<<1, 256, 0, stream>>>(YA, fcw, fcb, out);

  (void)in_sizes; (void)n_in; (void)out_size; (void)ws_size;
}
